// TestGraphNetwork_81827716923411
// MI455X (gfx1250) — compile-verified, matrix operand order checked by reading, not run on hardware
//
#include <hip/hip_runtime.h>

// ---------------------------------------------------------------------------
// Problem constants (match the reference)
// ---------------------------------------------------------------------------
#define NN      16384        // nodes
#define EE      131072       // edges (before self loops)
#define EP      (EE + NN)    // edges + self loops = 147456
#define GG      64           // graphs
#define IN_DIM  256
#define HID     128
#define HEADS   4
#define D2      (HEADS * HID)   // 512
#define GDIM    2
#define OUT_DIM 2

typedef __attribute__((ext_vector_type(16))) _Float16 v16h;
typedef __attribute__((ext_vector_type(8)))  _Float16 v8h;
typedef __attribute__((ext_vector_type(8)))  float    v8f;

// ---------------------------------------------------------------------------
// Helpers
// ---------------------------------------------------------------------------
__device__ __forceinline__ unsigned fordU(float f) {
    // monotone map float -> unsigned (for atomicMax-based segment max)
    unsigned u = __float_as_uint(f);
    return (u & 0x80000000u) ? ~u : (u | 0x80000000u);
}
__device__ __forceinline__ float unfordF(unsigned u) {
    unsigned v = (u & 0x80000000u) ? (u & 0x7FFFFFFFu) : ~u;
    return __uint_as_float(v);
}
#define FORD_NEG_INF 0x007FFFFFu   // fordU(-inf)

__device__ __forceinline__ float leaky(float v, float s) { return v > 0.0f ? v : s * v; }

// ---------------------------------------------------------------------------
// Fill kernel (init scratch: zeros / -inf sentinels)
// ---------------------------------------------------------------------------
__global__ void k_fill_u32(unsigned* __restrict__ p, unsigned val, long n) {
    long i = (long)blockIdx.x * blockDim.x + threadIdx.x;
    if (i < n) p[i] = val;
}

// ---------------------------------------------------------------------------
// fp32 -> fp16 elementwise (A-matrix prep)
// ---------------------------------------------------------------------------
__global__ void k_f32_to_f16(const float* __restrict__ src,
                             _Float16* __restrict__ dst, long n) {
    long i = (long)blockIdx.x * blockDim.x + threadIdx.x;
    if (i < n) dst[i] = (_Float16)src[i];
}

// ---------------------------------------------------------------------------
// Weight prep: W[K][N] fp32 -> Wt[N][K] fp16 (transposed so B fragments are
// contiguous b128 loads along K)
// ---------------------------------------------------------------------------
__global__ void k_w_to_f16t(const float* __restrict__ W,
                            _Float16* __restrict__ Wt, int K, int N) {
    long i = (long)blockIdx.x * blockDim.x + threadIdx.x;
    if (i >= (long)K * N) return;
    const int k = (int)(i / N);
    const int n = (int)(i - (long)k * N);
    Wt[(size_t)n * K + k] = (_Float16)W[i];
}

// ---------------------------------------------------------------------------
// WMMA GEMM:  C[M,Nn] = A[M,K] @ Bt[Nn,K]^T + bias[Nn]      (A, Bt in fp16)
//
// Block = 256 threads = 8 waves -> 128x64 output tile.
//   - 64x32 fp16 B^T k-chunk double-buffered in LDS (2x4KB); the global load
//     for chunk k+1 is ISSUED at the top of the iteration and only CONSUMED
//     (ds_store) after the 4 WMMAs, so its latency overlaps compute
//   - each wave: one 16-row A fragment (2x global b128) + 4 B fragments
//     (2x ds_load_b128 each) -> 4x v_wmma_f32_16x16x32_f16 per k-step
// ---------------------------------------------------------------------------
__global__ __launch_bounds__(256) void k_gemm_wmma(
    const _Float16* __restrict__ A, const _Float16* __restrict__ Bt,
    const float* __restrict__ bias, float* __restrict__ C,
    int M, int Nn, int K)
{
    __shared__ __align__(16) _Float16 Bs[2][64][32];   // [buf][col][k] fp16

    const int nBase = blockIdx.x * 64;
    const int mBase = blockIdx.y * 128;
    const int wave  = threadIdx.x >> 5;
    const int lane  = threadIdx.x & 31;
    const int hf    = lane >> 4;     // half-wave 0/1
    const int r     = lane & 15;

    // cooperative B-stage assignment: thread t loads 8 halves of row t/4
    const int sRow  = threadIdx.x >> 2;         // 0..63  (column index in block)
    const int sKoff = (threadIdx.x & 3) * 8;    // 0,8,16,24

    const int rowA = mBase + wave * 16 + r;
    const _Float16* Ar = A + (size_t)rowA * K;
    const _Float16* Br = Bt + (size_t)(nBase + sRow) * K + sKoff;

    // prologue: stage chunk 0 into buffer 0
    *(v8h*)&Bs[0][sRow][sKoff] = *(const v8h*)Br;

    v8f acc[4] = {};
    const int nk = K >> 5;
    for (int kc = 0; kc < nk; ++kc) {
        __syncthreads();                 // chunk kc visible; prev reads done
        const int cur = kc & 1;
        const int k0 = kc * 32;

        // Issue the stage load for chunk kc+1 NOW; consume it after the WMMAs
        // so its global latency is hidden behind compute.
        v8h stage = {};
        const bool doStage = (kc + 1 < nk);
        if (doStage) stage = *(const v8h*)(Br + (kc + 1) * 32);

        __builtin_prefetch(Ar + k0 + 128, 0, 3);   // near-cache prefetch (A stream)

        // A fragment: ISA 16-bit A 16x32 layout -> halves [hf*8..hf*8+7] then
        // [16+hf*8..16+hf*8+7], each an aligned contiguous b128 run.
        const v8h a0 = *(const v8h*)(Ar + k0 + hf * 8);
        const v8h a1 = *(const v8h*)(Ar + k0 + 16 + hf * 8);
        const v16h a = __builtin_shufflevector(a0, a1,
            0, 1, 2, 3, 4, 5, 6, 7, 8, 9, 10, 11, 12, 13, 14, 15);

        // load all 4 B fragments first, then issue WMMAs back-to-back
        v16h b[4];
#pragma unroll
        for (int j = 0; j < 4; ++j) {
            const v8h b0 = *(const v8h*)&Bs[cur][j * 16 + r][hf * 8];
            const v8h b1 = *(const v8h*)&Bs[cur][j * 16 + r][16 + hf * 8];
            b[j] = __builtin_shufflevector(b0, b1,
                0, 1, 2, 3, 4, 5, 6, 7, 8, 9, 10, 11, 12, 13, 14, 15);
        }
#pragma unroll
        for (int j = 0; j < 4; ++j) {
            acc[j] = __builtin_amdgcn_wmma_f32_16x16x32_f16(
                /*neg_a=*/false, a, /*neg_b=*/false, b[j],
                /*c_mod=*/(short)0, acc[j], /*reuse_a=*/false, /*reuse_b=*/false);
        }

        // Commit the staged chunk into the alternate buffer. Safe: all reads
        // of Bs[cur^1] happened before this iteration's barrier, and the
        // ds_store completes before the next barrier's dscnt wait.
        if (doStage) *(v8h*)&Bs[cur ^ 1][sRow][sKoff] = stage;
    }

    // C/D layout: VGPR jj -> row = tile_m + jj + 8*hf, col = tile_n + r
#pragma unroll
    for (int j = 0; j < 4; ++j) {
        const int col = nBase + j * 16 + r;
        const float bv = bias[col];
#pragma unroll
        for (int jj = 0; jj < 8; ++jj) {
            const int row = mBase + wave * 16 + jj + hf * 8;
            C[(size_t)row * Nn + col] = acc[j][jj] + bv;
        }
    }
}

// ---------------------------------------------------------------------------
// Edge scores: e[w,h] = att_h . leaky_relu(xl[src] + xr[dst], 0.2)
// one wave per edge; float4 lanes (wave covers 512B contiguous per step);
// wave-reduce per head; atomicMax segment max on dst.
// ---------------------------------------------------------------------------
__global__ __launch_bounds__(256) void k_edge_scores(
    const int* __restrict__ ei, const float* __restrict__ xl,
    const float* __restrict__ xr, const float* __restrict__ att,
    float* __restrict__ e, unsigned* __restrict__ mmax, int H, int CH)
{
    const int w = blockIdx.x * (blockDim.x >> 5) + (threadIdx.x >> 5);
    if (w >= EP) return;
    const int lane = threadIdx.x & 31;
    const int src = (w < EE) ? ei[w]      : (w - EE);
    const int dst = (w < EE) ? ei[EE + w] : (w - EE);
    const int HC = H * CH;

    float ph[HEADS] = {0.f, 0.f, 0.f, 0.f};
    for (int c0 = lane * 4; c0 < HC; c0 += 128) {
        const float4 l  = *(const float4*)(xl  + (size_t)src * HC + c0);
        const float4 rr = *(const float4*)(xr  + (size_t)dst * HC + c0);
        const float4 av = *(const float4*)(att + c0);
        // 4-channel group never crosses a head boundary (CH=128, c0%128<=124)
        ph[c0 / CH] += leaky(l.x + rr.x, 0.2f) * av.x
                     + leaky(l.y + rr.y, 0.2f) * av.y
                     + leaky(l.z + rr.z, 0.2f) * av.z
                     + leaky(l.w + rr.w, 0.2f) * av.w;
    }
    for (int h = 0; h < H; ++h) {
        float s = ph[h];
        for (int o = 16; o; o >>= 1) s += __shfl_xor(s, o, 32);
        if (lane == 0) {
            e[(size_t)w * H + h] = s;
            atomicMax(mmax + (size_t)dst * H + h, fordU(s));
        }
    }
}

// ---------------------------------------------------------------------------
// p = exp(e - m[dst]); denom[dst] += p   (thread per edge-head)
// ---------------------------------------------------------------------------
__global__ void k_edge_exp(
    const int* __restrict__ ei, float* __restrict__ e,
    const unsigned* __restrict__ mmax, float* __restrict__ denom, int H)
{
    const long i = (long)blockIdx.x * blockDim.x + threadIdx.x;
    if (i >= (long)EP * H) return;
    const int edge = (int)(i / H);
    const int h    = (int)(i - (long)edge * H);
    const int dst  = (edge < EE) ? ei[EE + edge] : (edge - EE);
    const float p  = __expf(e[i] - unfordF(mmax[(size_t)dst * H + h]));
    e[i] = p;
    atomicAdd(denom + (size_t)dst * H + h, p);
}

// ---------------------------------------------------------------------------
// out[dst] += (p/denom[dst]) * xl[src]   (wave per edge, float4 gather +
// scalar scatter-add)
// ---------------------------------------------------------------------------
__global__ __launch_bounds__(256) void k_edge_agg(
    const int* __restrict__ ei, const float* __restrict__ xl,
    const float* __restrict__ p, const float* __restrict__ denom,
    float* __restrict__ out, int H, int CH)
{
    const int w = blockIdx.x * (blockDim.x >> 5) + (threadIdx.x >> 5);
    if (w >= EP) return;
    const int lane = threadIdx.x & 31;
    const int src = (w < EE) ? ei[w]      : (w - EE);
    const int dst = (w < EE) ? ei[EE + w] : (w - EE);
    const int HC = H * CH;
    for (int c0 = lane * 4; c0 < HC; c0 += 128) {
        const int h = c0 / CH;
        const float alpha = p[(size_t)w * H + h] / denom[(size_t)dst * H + h];
        const float4 l = *(const float4*)(xl + (size_t)src * HC + c0);
        float* o = out + (size_t)dst * HC + c0;
        atomicAdd(o + 0, alpha * l.x);
        atomicAdd(o + 1, alpha * l.y);
        atomicAdd(o + 2, alpha * l.z);
        atomicAdd(o + 3, alpha * l.w);
    }
}

// ---------------------------------------------------------------------------
// h = leaky_relu(h + bias, 0.01)
// ---------------------------------------------------------------------------
__global__ void k_post(float* __restrict__ h, const float* __restrict__ bias,
                       int HC, long n)
{
    const long i = (long)blockIdx.x * blockDim.x + threadIdx.x;
    if (i >= n) return;
    h[i] = leaky(h[i] + bias[i % HC], 0.01f);
}

// ---------------------------------------------------------------------------
// Global mean pool (scatter phase)
// ---------------------------------------------------------------------------
__global__ void k_pool(const float* __restrict__ h, const int* __restrict__ batch,
                       float* __restrict__ pooled, float* __restrict__ counts)
{
    const long i = (long)blockIdx.x * blockDim.x + threadIdx.x;
    if (i >= (long)NN * HID) return;
    const int n = (int)(i / HID);
    const int c = (int)(i - (long)n * HID);
    const int g = batch[n];
    atomicAdd(pooled + (size_t)g * HID + c, h[i]);
    if (c == 0) atomicAdd(counts + g, 1.0f);
}

// ---------------------------------------------------------------------------
// Classifier head: z = [pooled/cnt, gf]; out = leaky(z W1 + b1) W2 + b2
// One block (128 threads) per graph; tiny, scalar is fine.
// ---------------------------------------------------------------------------
__global__ __launch_bounds__(128) void k_classifier(
    const float* __restrict__ pooled, const float* __restrict__ counts,
    const float* __restrict__ gf,
    const float* __restrict__ W1, const float* __restrict__ b1,
    const float* __restrict__ W2, const float* __restrict__ b2,
    float* __restrict__ out)
{
    const int g = blockIdx.x;
    const int j = threadIdx.x;          // 0..127
    __shared__ float z[HID + GDIM];
    __shared__ float z1[HID];

    z[j] = pooled[(size_t)g * HID + j] / fmaxf(counts[g], 1.0f);
    if (j < GDIM) z[HID + j] = gf[(size_t)g * GDIM + j];
    __syncthreads();

    float s = b1[j];
    for (int i = 0; i < HID + GDIM; ++i) s += z[i] * W1[(size_t)i * HID + j];
    z1[j] = leaky(s, 0.01f);
    __syncthreads();

    if (j < OUT_DIM) {
        float o = b2[j];
        for (int k = 0; k < HID; ++k) o += z1[k] * W2[(size_t)k * OUT_DIM + j];
        out[(size_t)g * OUT_DIM + j] = o;
    }
}

// ---------------------------------------------------------------------------
// Host orchestration
// ---------------------------------------------------------------------------
static inline void fill_u32(unsigned* p, unsigned v, long n, hipStream_t s) {
    const int T = 256;
    k_fill_u32<<<(int)((n + T - 1) / T), T, 0, s>>>(p, v, n);
}

extern "C" void kernel_launch(void* const* d_in, const int* in_sizes, int n_in,
                              void* d_out, int out_size, void* d_ws, size_t ws_size,
                              hipStream_t stream) {
    (void)in_sizes; (void)n_in; (void)out_size; (void)ws_size;

    const float* x     = (const float*)d_in[0];
    const int*   ei    = (const int*)  d_in[1];
    const int*   batch = (const int*)  d_in[2];
    const float* gf    = (const float*)d_in[3];

    const float* c_Wl[3]   = {(const float*)d_in[4],  (const float*)d_in[10], (const float*)d_in[16]};
    const float* c_bl[3]   = {(const float*)d_in[5],  (const float*)d_in[11], (const float*)d_in[17]};
    const float* c_Wr[3]   = {(const float*)d_in[6],  (const float*)d_in[12], (const float*)d_in[18]};
    const float* c_br[3]   = {(const float*)d_in[7],  (const float*)d_in[13], (const float*)d_in[19]};
    const float* c_att[3]  = {(const float*)d_in[8],  (const float*)d_in[14], (const float*)d_in[20]};
    const float* c_bias[3] = {(const float*)d_in[9],  (const float*)d_in[15], (const float*)d_in[21]};
    const float* W1 = (const float*)d_in[22];
    const float* b1 = (const float*)d_in[23];
    const float* W2 = (const float*)d_in[24];
    const float* b2 = (const float*)d_in[25];
    float* out = (float*)d_out;

    // ---- workspace layout (256B aligned) ----
    char* ws = (char*)d_ws;
    size_t off = 0;
    auto carve = [&](size_t bytes) -> char* {
        char* p = ws + off;
        off += (bytes + 255) & ~(size_t)255;
        return p;
    };
    float*    bufA   = (float*)   carve((size_t)NN * D2 * 4);  // node buffer (aliased in/out via stream order)
    float*    xl     = (float*)   carve((size_t)NN * D2 * 4);
    float*    xr     = (float*)   carve((size_t)NN * D2 * 4);
    _Float16* Ah     = (_Float16*)carve((size_t)NN * D2 * 2);  // fp16 A
    _Float16* Wt     = (_Float16*)carve((size_t)D2 * D2 * 2);  // fp16 B^T (reused Wl/Wr)
    float*    ebuf   = (float*)   carve((size_t)EP * HEADS * 4);
    unsigned* mmax   = (unsigned*)carve((size_t)NN * HEADS * 4);
    float*    denom  = (float*)   carve((size_t)NN * HEADS * 4);
    float*    pooled = (float*)   carve((size_t)GG * HID * 4);
    float*    counts = (float*)   carve((size_t)GG * 4);

    const int T = 256;
    const int edgeBlocks = (EP + 7) / 8;   // 8 waves per 256-thread block

    // One GATv2 layer. `in` is read only by the fp16 conversion, so `outBuf`
    // may alias it (stream order: convert -> GEMMs -> zero -> scatter -> post).
    auto gat_layer = [&](const float* in, int K, int H, int CH, int li, float* outBuf) {
        const int HC = H * CH;
        const dim3 gemmGrid(HC / 64, NN / 128);

        k_f32_to_f16<<<(int)(((long)NN * K + T - 1) / T), T, 0, stream>>>(in, Ah, (long)NN * K);

        k_w_to_f16t<<<(int)(((long)K * HC + T - 1) / T), T, 0, stream>>>(c_Wl[li], Wt, K, HC);
        k_gemm_wmma<<<gemmGrid, T, 0, stream>>>(Ah, Wt, c_bl[li], xl, NN, HC, K);
        k_w_to_f16t<<<(int)(((long)K * HC + T - 1) / T), T, 0, stream>>>(c_Wr[li], Wt, K, HC);
        k_gemm_wmma<<<gemmGrid, T, 0, stream>>>(Ah, Wt, c_br[li], xr, NN, HC, K);

        fill_u32((unsigned*)outBuf, 0u, (long)NN * HC, stream);
        fill_u32(mmax, FORD_NEG_INF, (long)NN * H, stream);
        fill_u32((unsigned*)denom, 0u, (long)NN * H, stream);

        k_edge_scores<<<edgeBlocks, T, 0, stream>>>(ei, xl, xr, c_att[li], ebuf, mmax, H, CH);
        k_edge_exp<<<(int)(((long)EP * H + T - 1) / T), T, 0, stream>>>(ei, ebuf, mmax, denom, H);
        k_edge_agg<<<edgeBlocks, T, 0, stream>>>(ei, xl, ebuf, denom, outBuf, H, CH);
        k_post<<<(int)(((long)NN * HC + T - 1) / T), T, 0, stream>>>(outBuf, c_bias[li], HC, (long)NN * HC);
    };

    gat_layer(x,    IN_DIM, HEADS, HID, 0, bufA);   // 256 -> 512
    gat_layer(bufA, D2,     HEADS, HID, 1, bufA);   // 512 -> 512
    gat_layer(bufA, D2,     1,     HID, 2, bufA);   // 512 -> 128 (heads=1)

    fill_u32((unsigned*)pooled, 0u, (long)GG * HID, stream);
    fill_u32((unsigned*)counts, 0u, GG, stream);
    k_pool<<<(int)(((long)NN * HID + T - 1) / T), T, 0, stream>>>(bufA, batch, pooled, counts);

    k_classifier<<<GG, HID, 0, stream>>>(pooled, counts, gf, W1, b1, W2, b2, out);
}
